// info_nce_loss_18545668784128
// MI455X (gfx1250) — matrix operand order repaired and checked by reading, not compiled, not run on hardware
//
#include <hip/hip_runtime.h>
#include <hip/hip_bf16.h>
#include <math.h>

typedef __attribute__((ext_vector_type(16))) _Float16 v16h;
typedef __attribute__((ext_vector_type(8)))  _Float16 v8h;
typedef __attribute__((ext_vector_type(8)))  float    v8f;

#define BB    4
#define CC    128
#define HH    256
#define WW    256
#define DD    3000
#define DPAD  3008          // pad D to multiple of 16
#define NT    188           // DPAD / 16 tiles
#define NGRP  47            // NT / 4 waves per block  (exact)
#define INV_T 10.0f
#define EPSN  1e-12f
#define EXP_SCALE (10.0f * 1.4426950408889634f)   // INV_T * log2(e), for v_exp_f32

// ---------------------------------------------------------------------------
// Kernel 1: gather rows of pred (layout pred[b][c][h][w], spatial index
// s = w*H + h per the reference's permute), L2-normalize over C, store f16.
// One wave of 32 lanes per gathered row; 4 channels per lane.
// ---------------------------------------------------------------------------
__global__ void gather_norm_kernel(const float* __restrict__ pred,
                                   const int*   __restrict__ q_idx,
                                   const int*   __restrict__ n_idx,
                                   _Float16*    __restrict__ q_h,
                                   _Float16*    __restrict__ n_h) {
    int row  = blockIdx.x * 8 + (threadIdx.x >> 5);
    int lane = threadIdx.x & 31;
    int b     = row / (2 * DPAD);
    int rem   = row - b * (2 * DPAD);
    int which = rem / DPAD;          // 0 = q, 1 = n
    int d     = rem - which * DPAD;

    _Float16* out = (which ? n_h : q_h) + ((size_t)b * DPAD + d) * CC;
    if (d >= DD) {                    // zero-pad rows so they contribute nothing
        #pragma unroll
        for (int k = 0; k < 4; ++k) out[lane + 32 * k] = (_Float16)0.0f;
        return;
    }
    const int* idx = which ? n_idx : q_idx;
    int s = idx[b * DD + d];
    int w = s / HH;                   // s = w*H + h
    int h = s - w * HH;

    float v[4];
    float ss = 0.0f;
    #pragma unroll
    for (int k = 0; k < 4; ++k) {
        int c = lane + 32 * k;
        v[k] = pred[(((size_t)b * CC + c) * HH + h) * WW + w];
        ss += v[k] * v[k];
    }
    #pragma unroll
    for (int m = 16; m >= 1; m >>= 1) ss += __shfl_xor(ss, m, 32);
    float inv = 1.0f / fmaxf(sqrtf(ss), EPSN);
    #pragma unroll
    for (int k = 0; k < 4; ++k) out[lane + 32 * k] = (_Float16)(v[k] * inv);
}

// ---------------------------------------------------------------------------
// Helpers: LDS offset of a __shared__ pointer, and the CDNA5 async
// global->LDS copy (tracked by ASYNCcnt; INST_OFFSET applies to both sides).
// ---------------------------------------------------------------------------
typedef __attribute__((address_space(3))) const void lds_cv_t;

__device__ __forceinline__ unsigned lds_offset(const void* p) {
    return (unsigned)(unsigned long long)(lds_cv_t*)p;
}

__device__ __forceinline__ void async_copy32(const _Float16* gsrc, unsigned ldst) {
    // two b128 ops cover one 32-byte swizzled chunk
    asm volatile("global_load_async_to_lds_b128 %0, %1, off"
                 :: "v"(ldst), "v"(gsrc) : "memory");
    asm volatile("global_load_async_to_lds_b128 %0, %1, off offset:16"
                 :: "v"(ldst), "v"(gsrc) : "memory");
}

__device__ __forceinline__ void wait_async0() {
    asm volatile("s_wait_asynccnt 0x0" ::: "memory");
}

// ---------------------------------------------------------------------------
// Kernel 2: block = 4 waves, each wave owns one 16-row q i-tile. The block
// loops over all 188 j-tiles; each j-tile (q rows + n rows, 4 KB each) is
// staged in LDS (ping-pong, XOR-swizzled 32 B chunks) via the CDNA5 async
// global->LDS path, overlapping the 8 WMMAs on the current tile. One
// s_wait_asynccnt + barrier per iteration.
// ---------------------------------------------------------------------------
__global__ void nce_tile_kernel(const _Float16* __restrict__ q_h,
                                const _Float16* __restrict__ n_h,
                                float*          __restrict__ out) {
    __shared__ alignas(32) _Float16 ldsQ[2][16 * CC];   // 2 x 4 KB
    __shared__ alignas(32) _Float16 ldsN[2][16 * CC];   // 2 x 4 KB
    __shared__ float sS[4][16], sE[4][16], sDg[4][16];

    int tid  = threadIdx.x;
    int b    = blockIdx.x / NGRP;
    int grp  = blockIdx.x - b * NGRP;
    int wave = tid >> 5;
    int lane = tid & 31;
    int g    = lane >> 4;     // lane group (0/1)
    int nc   = lane & 15;     // row (for A loads) / column (for C frag)
    int it   = grp * 4 + wave;            // this wave's i-tile, always < NT
    int i0   = it * 16;

    const _Float16* qb = q_h + (size_t)b * DPAD * CC;
    const _Float16* nb = n_h + (size_t)b * DPAD * CC;

    // A fragments (16x32 f16): lane holds row M=nc; per K-chunk kk, halves
    // [16g .. 16g+15] of that row. Loaded once from global.
    const _Float16* arow = qb + (size_t)(i0 + nc) * CC;
    v16h afrag[4];
    #pragma unroll
    for (int kk = 0; kk < 4; ++kk) {
        const _Float16* p = arow + kk * 32 + g * 16;
        v8h lo = *(const v8h*)(p);
        v8h hi = *(const v8h*)(p + 8);
        v16h t;
        #pragma unroll
        for (int i = 0; i < 8; ++i) { t[i] = lo[i]; t[8 + i] = hi[i]; }
        afrag[kk] = t;
    }

    float accS[8], accE[8], accDg[8];
    #pragma unroll
    for (int v = 0; v < 8; ++v) { accS[v] = 0.0f; accE[v] = 0.0f; accDg[v] = 0.0f; }

    // Staging geometry: thread copies one 32 B chunk. 16 rows x 8 chunks.
    int srow = tid >> 3;                  // 0..15
    int schk = tid & 7;                   // 0..7
    int ssw  = schk ^ (srow & 7);         // XOR swizzle target chunk
    unsigned dstQ[2], dstN[2];            // LDS byte offsets, per buffer
    dstQ[0] = lds_offset(&ldsQ[0][srow * CC + ssw * 16]);
    dstQ[1] = lds_offset(&ldsQ[1][srow * CC + ssw * 16]);
    dstN[0] = lds_offset(&ldsN[0][srow * CC + ssw * 16]);
    dstN[1] = lds_offset(&ldsN[1][srow * CC + ssw * 16]);

    // Prologue: stage tile 0 into buffer 0 (async), wait, barrier.
    async_copy32(qb + (size_t)srow * CC + schk * 16, dstQ[0]);
    async_copy32(nb + (size_t)srow * CC + schk * 16, dstN[0]);
    wait_async0();
    __syncthreads();

    for (int jt = 0; jt < NT; ++jt) {
        int cur = jt & 1;
        int j0  = jt * 16;

        // Issue next tile's async global->LDS copies (overlap with WMMAs).
        bool hasNext = (jt + 1 < NT);
        if (hasNext) {
            int nxt = cur ^ 1;
            size_t roff = (size_t)((jt + 1) * 16 + srow) * CC + schk * 16;
            async_copy32(qb + roff, dstQ[nxt]);
            async_copy32(nb + roff, dstN[nxt]);
        }

        // B fragments from LDS (de-swizzled); 8 WMMAs, two independent chains.
        const _Float16* lq = &ldsQ[cur][nc * CC];
        const _Float16* ln = &ldsN[cur][nc * CC];
        v8f csim = {};
        v8f cneg = {};
        #pragma unroll
        for (int kk = 0; kk < 4; ++kk) {
            int swc = ((kk * 2 + g) ^ (nc & 7)) * 16;
            v16h bq = *(const v16h*)(lq + swc);
            v16h bn = *(const v16h*)(ln + swc);
            csim = __builtin_amdgcn_wmma_f32_16x16x32_f16(
                false, afrag[kk], false, bq, (short)0, csim, false, false);
            cneg = __builtin_amdgcn_wmma_f32_16x16x32_f16(
                false, afrag[kk], false, bn, (short)0, cneg, false, false);
        }

        #pragma unroll
        for (int v = 0; v < 8; ++v) accS[v] += csim[v];   // padded cols exact 0
        if (j0 == i0) {                                   // diagonal tile (rare)
            #pragma unroll
            for (int v = 0; v < 8; ++v)
                if (nc == v + 8 * g) accDg[v] += csim[v];
        }
        float mvalid = ((j0 + nc) < DD) ? 1.0f : 0.0f;    // mask padded n cols
        #pragma unroll
        for (int v = 0; v < 8; ++v)
            accE[v] += mvalid * __builtin_amdgcn_exp2f(cneg[v] * EXP_SCALE);

        // Async copies must land before the next iteration reads them.
        wait_async0();
        __syncthreads();
    }

    // Per-wave reduction across the 16 column-lanes (wave32 xor shuffles).
    #pragma unroll
    for (int v = 0; v < 8; ++v) {
        float s = accS[v], e = accE[v], dg = accDg[v];
        #pragma unroll
        for (int m = 1; m <= 8; m <<= 1) {
            s  += __shfl_xor(s,  m, 32);
            e  += __shfl_xor(e,  m, 32);
            dg += __shfl_xor(dg, m, 32);
        }
        if (nc == 0) {                  // lanes 0 and 16 hold rows v and v+8
            int r = v + 8 * g;
            sS[wave][r]  = s;           // single writer per slot: plain stores
            sE[wave][r]  = e;
            sDg[wave][r] = dg;
        }
    }
    __syncthreads();

    if (tid < 64) {
        int wv = tid >> 4;
        int r  = tid & 15;
        int i  = (grp * 4 + wv) * 16 + r;
        if (i < DD) {
            float pos  = (sS[wv][r] - sDg[wv][r]) * (1.0f / (float)(DD - 1));
            float l0   = pos * INV_T;
            // dots bounded in [-1,1] => logits/T in [-10,10]: safe without max-sub
            float loss = logf(expf(l0) + sE[wv][r]) - l0;
            atomicAdd(out, loss * (1.0f / ((float)BB * (float)DD)));
        }
    }
}

__global__ void zero_kernel(float* out) {
    if (threadIdx.x == 0 && blockIdx.x == 0) *out = 0.0f;
}

extern "C" void kernel_launch(void* const* d_in, const int* in_sizes, int n_in,
                              void* d_out, int out_size, void* d_ws, size_t ws_size,
                              hipStream_t stream) {
    const float* pred  = (const float*)d_in[0];
    // d_in[1] = target (unused: q_idx/n_idx already encode the selection)
    const int*   q_idx = (const int*)d_in[2];
    const int*   n_idx = (const int*)d_in[3];
    float*       out   = (float*)d_out;

    _Float16* q_h = (_Float16*)d_ws;                       // [B][DPAD][C] f16
    _Float16* n_h = q_h + (size_t)BB * DPAD * CC;          // total ~6.2 MB

    int rows = BB * 2 * DPAD;                              // 24064 gathered rows
    gather_norm_kernel<<<rows / 8, 256, 0, stream>>>(pred, q_idx, n_idx, q_h, n_h);
    zero_kernel<<<1, 32, 0, stream>>>(out);
    nce_tile_kernel<<<BB * NGRP, 128, 0, stream>>>(q_h, n_h, out);
}
